// LlamaAttention_59648505806915
// MI455X (gfx1250) — matrix operand order repaired and checked by reading, not compiled, not run on hardware
//
#include <hip/hip_runtime.h>
#include <cstdint>
#include <cstddef>

typedef __attribute__((ext_vector_type(16))) __bf16 v16bf;
typedef __attribute__((ext_vector_type(8)))  __bf16 v8bf;
typedef __attribute__((ext_vector_type(4)))  __bf16 v4bf;
typedef __attribute__((ext_vector_type(8)))  float  v8f;

#define WMMA_BF16(a, b, c) \
  __builtin_amdgcn_wmma_f32_16x16x32_bf16(false, (a), false, (b), (short)0, (c), false, false)

// Problem constants
constexpr int Bn   = 2;
constexpr int S    = 1024;
constexpr int Pc   = 1024;           // past tokens
constexpr int Tt   = 2048;           // total tokens
constexpr int H    = 4096;
constexpr int NQ   = 32;
constexpr int NKV  = 8;
constexpr int D    = 128;
constexpr int Mrows = Bn * S;        // 2048
constexpr int NQKV  = (NQ + 2 * NKV) * D;  // 6144
constexpr float SCALE = 0.08838834764831845f;  // 1/sqrt(128)

// ---------- fragment loaders (wave32 WMMA layouts, ISA 7.12.2) ----------
__device__ __forceinline__ v16bf frag_contig(const __bf16* p) {
  v8bf lo = *(const v8bf*)p;
  v8bf hi = *(const v8bf*)(p + 8);
  return __builtin_shufflevector(lo, hi, 0,1,2,3,4,5,6,7,8,9,10,11,12,13,14,15);
}
__device__ __forceinline__ v16bf frag_split(const __bf16* p) {
  v8bf lo = *(const v8bf*)p;
  v8bf hi = *(const v8bf*)(p + 16);
  return __builtin_shufflevector(lo, hi, 0,1,2,3,4,5,6,7,8,9,10,11,12,13,14,15);
}

// ---------- VALU (DPP) all-reduce within each 16-lane half ----------
// row_ror:N rotates within a 16-lane DPP row; ror 1,2,4,8 -> full all-reduce.
template <int CTRL>
__device__ __forceinline__ float dpp_mov(float v) {
  return __int_as_float(
      __builtin_amdgcn_update_dpp(0, __float_as_int(v), CTRL, 0xf, 0xf, true));
}
__device__ __forceinline__ float red16_max(float v) {
  v = fmaxf(v, dpp_mov<0x121>(v));   // row_ror:1
  v = fmaxf(v, dpp_mov<0x122>(v));   // row_ror:2
  v = fmaxf(v, dpp_mov<0x124>(v));   // row_ror:4
  v = fmaxf(v, dpp_mov<0x128>(v));   // row_ror:8
  return v;
}
__device__ __forceinline__ float red16_sum(float v) {
  v += dpp_mov<0x121>(v);
  v += dpp_mov<0x122>(v);
  v += dpp_mov<0x124>(v);
  v += dpp_mov<0x128>(v);
  return v;
}

// ---------- async global -> LDS staging (ASYNCcnt path), with fallback ----------
#if defined(__HIP_DEVICE_COMPILE__) && __has_builtin(__builtin_amdgcn_global_load_async_to_lds_b128)
#define HAS_ASYNC_LDS 1
#else
#define HAS_ASYNC_LDS 0
#endif

typedef __attribute__((ext_vector_type(4))) int i4v;
typedef __attribute__((address_space(1))) i4v* gp_i4;   // int4 in global (AS1)
typedef __attribute__((address_space(3))) i4v* lp_i4;   // int4 in LDS (AS3)

__device__ __forceinline__ void stage_b128(__bf16* lds_dst, const __bf16* gsrc) {
#if HAS_ASYNC_LDS
  __builtin_amdgcn_global_load_async_to_lds_b128((gp_i4)gsrc, (lp_i4)lds_dst, 0, 0);
#else
  *(v8bf*)lds_dst = *(const v8bf*)gsrc;
#endif
}
// wait until at most the most recent 8 async ops remain outstanding
__device__ __forceinline__ void stage_wait_prev() {
#if HAS_ASYNC_LDS
  asm volatile("s_wait_asynccnt 0x8" ::: "memory");
#endif
}
__device__ __forceinline__ void stage_wait_all() {
#if HAS_ASYNC_LDS
  asm volatile("s_wait_asynccnt 0x0" ::: "memory");
#endif
}

// ---------- f32 -> bf16 convert (4 elems/thread) ----------
__global__ void k_cvt_bf16(const float* __restrict__ src, __bf16* __restrict__ dst, int n) {
  int i = (blockIdx.x * blockDim.x + threadIdx.x) * 4;
  if (i < n) {
    float4 f = *(const float4*)(src + i);
    v4bf o;
    o[0] = (__bf16)f.x; o[1] = (__bf16)f.y; o[2] = (__bf16)f.z; o[3] = (__bf16)f.w;
    *(v4bf*)(dst + i) = o;
  }
}

// ---------- transpose + convert: src[K][N] f32 -> dst[N][K] bf16 ----------
__global__ void k_transpose_cvt(const float* __restrict__ src, __bf16* __restrict__ dst,
                                int K, int N) {
  __shared__ float tile[32][33];
  int x0 = blockIdx.x * 32;   // N
  int y0 = blockIdx.y * 32;   // K
  int tx = threadIdx.x, ty = threadIdx.y;
#pragma unroll
  for (int i = 0; i < 4; ++i)
    tile[ty + i * 8][tx] = src[(size_t)(y0 + ty + i * 8) * N + x0 + tx];
  __syncthreads();
#pragma unroll
  for (int i = 0; i < 4; ++i)
    dst[(size_t)(x0 + ty + i * 8) * K + y0 + tx] = (__bf16)tile[tx][ty + i * 8];
}

// ---------- KV cache convert/scatter ----------
__global__ void k_cvt_kv(const float* __restrict__ kc, const float* __restrict__ vc,
                         __bf16* __restrict__ kall, __bf16* __restrict__ vallT) {
  int i = blockIdx.x * blockDim.x + threadIdx.x;   // 0 .. B*P*NKV*D-1
  int d = i & 127;
  int n = (i >> 7) & 7;
  int p = (i >> 10) & 1023;
  int b = i >> 20;
  kall[((size_t)(b * 8 + n) * 2048 + p) * 128 + d] = (__bf16)kc[i];
  vallT[((size_t)(b * 8 + n) * 128 + d) * 2048 + p] = (__bf16)vc[i];
}

// ---------- WMMA GEMM: C[M][N] = A[M][K] * Bt[N][K]^T ----------
template <int OUT_BF16>
__global__ __launch_bounds__(256) void k_gemm(const __bf16* __restrict__ A,
                                              const __bf16* __restrict__ Bt,
                                              void* __restrict__ C,
                                              int M, int N, int K) {
  const int lane = threadIdx.x & 31;
  const int wave = threadIdx.x >> 5;
  const int half = lane >> 4;
  const int wm = blockIdx.y * 128 + (wave >> 1) * 32;
  const int wn = blockIdx.x * 128 + (wave & 1) * 64;

  v8f zero = {};
  v8f acc[2][4];
#pragma unroll
  for (int mt = 0; mt < 2; ++mt)
#pragma unroll
    for (int nt = 0; nt < 4; ++nt) acc[mt][nt] = zero;

  const __bf16* ar[2];
  const __bf16* br[4];
#pragma unroll
  for (int mt = 0; mt < 2; ++mt)
    ar[mt] = A + (size_t)(wm + mt * 16 + (lane & 15)) * K + half * 16;
#pragma unroll
  for (int nt = 0; nt < 4; ++nt)
    br[nt] = Bt + (size_t)(wn + nt * 16 + (lane & 15)) * K + half * 16;

  for (int kk = 0; kk < K; kk += 32) {
    v16bf af[2], bf[4];
#pragma unroll
    for (int mt = 0; mt < 2; ++mt) af[mt] = frag_contig(ar[mt] + kk);
#pragma unroll
    for (int nt = 0; nt < 4; ++nt) bf[nt] = frag_contig(br[nt] + kk);
#pragma unroll
    for (int mt = 0; mt < 2; ++mt)
#pragma unroll
      for (int nt = 0; nt < 4; ++nt)
        acc[mt][nt] = WMMA_BF16(af[mt], bf[nt], acc[mt][nt]);
  }

#pragma unroll
  for (int mt = 0; mt < 2; ++mt)
#pragma unroll
    for (int nt = 0; nt < 4; ++nt)
#pragma unroll
      for (int j = 0; j < 8; ++j) {
        int row = wm + mt * 16 + j + half * 8;
        int col = wn + nt * 16 + (lane & 15);
        if (OUT_BF16)
          ((__bf16*)C)[(size_t)row * N + col] = (__bf16)acc[mt][nt][j];
        else
          ((float*)C)[(size_t)row * N + col] = acc[mt][nt][j];
      }
}

// ---------- RoPE + scatter ----------
__global__ void k_rope_scatter(const __bf16* __restrict__ qkv,
                               const float* __restrict__ cosb, const float* __restrict__ sinb,
                               __bf16* __restrict__ qb, __bf16* __restrict__ kall,
                               __bf16* __restrict__ vallT) {
  int m = blockIdx.x;          // 0..2047
  int hh = blockIdx.y;         // 0..47
  int d = threadIdx.x;         // 0..127
  int b = m >> 10, s = m & 1023;
  const __bf16* row = qkv + (size_t)m * NQKV + hh * 128;
  float x = (float)row[d];
  if (hh >= 40) {  // V: plain copy, transposed layout
    int n = hh - 40;
    vallT[((size_t)(b * 8 + n) * 128 + d) * 2048 + Pc + s] = (__bf16)x;
    return;
  }
  float rot = (d < 64) ? -(float)row[d + 64] : (float)row[d - 64];
  float val = x * cosb[s * 128 + d] + rot * sinb[s * 128 + d];
  if (hh < 32) {
    qb[((size_t)(b * 32 + hh) * 1024 + s) * 128 + d] = (__bf16)(val * SCALE);
  } else {
    int n = hh - 32;
    kall[((size_t)(b * 8 + n) * 2048 + Pc + s) * 128 + d] = (__bf16)val;
  }
}

// ---------- Flash attention with cooperative async K/V staging ----------
// block = (b, kv head n, pair of s-tiles); 8 waves = 4 GQA heads x 2 s-tiles,
// all sharing the same K/V stream. K/V tiles double-buffered in LDS, staged
// with GLOBAL_LOAD_ASYNC_TO_LDS_B128 (prefetch next t-block during compute).
__global__ __launch_bounds__(256) void k_attn(const __bf16* __restrict__ qb,
                                              const __bf16* __restrict__ kall,
                                              const __bf16* __restrict__ vallT,
                                              __bf16* __restrict__ attnb) {
  __shared__ __bf16 kst[2][64 * 128];   // [t-t0][d]      16KB x2
  __shared__ __bf16 vst[2][128 * 64];   // [d][t-t0]      16KB x2
  __shared__ __bf16 plds[8 * 16 * 64];  // per-wave P staging, 16KB

  const int tid = threadIdx.x;
  const int lane = tid & 31;
  const int wave = tid >> 5;
  const int half = lane >> 4;

  // block decode: 512 blocks = B(2) x NKV(8) x 32 tile-pairs
  const int b = blockIdx.x >> 8;
  const int rem = blockIdx.x & 255;
  const int n = rem >> 5;              // kv head
  const int pi = rem & 31;             // s-tile pair index
  const int g = wave >> 1;             // GQA sub-head
  const int st = pi * 2 + (wave & 1);  // s-tile of this wave
  const int h = n * 4 + g;             // q head
  const int s0 = st * 16;

  __bf16* pl = plds + wave * 1024;
  const __bf16* kbase = kall + (size_t)(b * 8 + n) * 2048 * 128;
  const __bf16* vbase = vallT + (size_t)(b * 8 + n) * 128 * 2048;

  // Q fragments (A operand), scaled by 1/sqrt(D) already
  const __bf16* qrow = qb + ((size_t)(b * 32 + h) * 1024 + s0 + (lane & 15)) * 128 + half * 16;
  v16bf aq[4];
#pragma unroll
  for (int kk = 0; kk < 4; ++kk) aq[kk] = frag_contig(qrow + kk * 32);

  v8f zero = {};
  v8f o[8];
#pragma unroll
  for (int dt = 0; dt < 8; ++dt) o[dt] = zero;
  float mrun[8], lrun[8];
#pragma unroll
  for (int j = 0; j < 8; ++j) { mrun[j] = -3.0e38f; lrun[j] = 0.0f; }

  // cooperative stage of one 64-token K/V block (8 async b128 per thread)
  auto issue_stage = [&](int t0, int buf) {
#pragma unroll
    for (int it = 0; it < 4; ++it) {
      int o8 = (tid + it * 256) * 8;                 // element offset, 16B chunks
      stage_b128(&kst[buf][o8], kbase + (size_t)t0 * 128 + o8);
      int dr = o8 >> 6;                              // V row (d), 64 elems/row
      int ir = o8 & 63;
      stage_b128(&vst[buf][o8], vbase + (size_t)dr * 2048 + t0 + ir);
    }
  };

  const int blk_tmax = (pi * 2 + 1) * 16 + 15 + Pc;  // block-max valid t (inclusive)
  issue_stage(0, 0);

  for (int t0 = 0; t0 <= blk_tmax; t0 += 64) {
    const int buf = (t0 >> 6) & 1;
    const bool hasNext = (t0 + 64) <= blk_tmax;
    if (hasNext) {
      issue_stage(t0 + 64, buf ^ 1);   // prefetch next block into other buffer
      stage_wait_prev();               // current buffer's 8 ops retired
    } else {
      stage_wait_all();
    }
    __syncthreads();                   // all waves' staged data visible

    // ---- scores S = Q * K^T over 64 columns (K from LDS) ----
    v8f sc[4];
#pragma unroll
    for (int nt = 0; nt < 4; ++nt) sc[nt] = zero;
#pragma unroll
    for (int kk = 0; kk < 4; ++kk) {
#pragma unroll
      for (int nt = 0; nt < 4; ++nt) {
        const __bf16* kp = &kst[buf][(nt * 16 + (lane & 15)) * 128 + kk * 32 + half * 16];
        sc[nt] = WMMA_BF16(aq[kk], frag_contig(kp), sc[nt]);
      }
    }
    // ---- causal mask: t <= s + P (fully-masked blocks are numerically inert) ----
    if (t0 + 63 > s0 + Pc) {
#pragma unroll
      for (int nt = 0; nt < 4; ++nt)
#pragma unroll
        for (int j = 0; j < 8; ++j) {
          int te = t0 + nt * 16 + (lane & 15);
          int se = s0 + j + half * 8 + Pc;
          if (te > se) sc[nt][j] = -3.0e38f;
        }
    }
    // ---- online softmax (row reductions on VALU via DPP row_ror) ----
    float alpha[8];
#pragma unroll
    for (int j = 0; j < 8; ++j) {
      float mx = fmaxf(fmaxf(sc[0][j], sc[1][j]), fmaxf(sc[2][j], sc[3][j]));
      mx = red16_max(mx);
      float mn = fmaxf(mrun[j], mx);
      alpha[j] = __expf(mrun[j] - mn);
      mrun[j] = mn;
    }
    float rs[8];
#pragma unroll
    for (int j = 0; j < 8; ++j) rs[j] = 0.0f;
#pragma unroll
    for (int nt = 0; nt < 4; ++nt)
#pragma unroll
      for (int j = 0; j < 8; ++j) {
        float pv = __expf(sc[nt][j] - mrun[j]);
        sc[nt][j] = pv;
        rs[j] += pv;
      }
#pragma unroll
    for (int j = 0; j < 8; ++j) {
      rs[j] = red16_sum(rs[j]);
      lrun[j] = lrun[j] * alpha[j] + rs[j];
    }
#pragma unroll
    for (int dt = 0; dt < 8; ++dt)
#pragma unroll
      for (int j = 0; j < 8; ++j) o[dt][j] *= alpha[j];

    // ---- stage P (C-layout -> row-major LDS), reload as A fragments ----
#pragma unroll
    for (int nt = 0; nt < 4; ++nt)
#pragma unroll
      for (int j = 0; j < 8; ++j)
        pl[(j + half * 8) * 64 + nt * 16 + (lane & 15)] = (__bf16)sc[nt][j];
    asm volatile("s_wait_dscnt 0x0" ::: "memory");

#pragma unroll
    for (int kt = 0; kt < 2; ++kt) {
      v16bf pf = frag_contig(pl + (lane & 15) * 64 + kt * 32 + half * 16);
#pragma unroll
      for (int dt = 0; dt < 8; ++dt) {
        const __bf16* vp = &vst[buf][(dt * 16 + (lane & 15)) * 64 + kt * 32 + half * 16];
        o[dt] = WMMA_BF16(pf, frag_contig(vp), o[dt]);
      }
    }
    __syncthreads();   // everyone done reading buf before it is overwritten
  }

  // ---- finalize: O /= rowsum, store to attnb [b][s][h*128+d] ----
  float inv[8];
#pragma unroll
  for (int j = 0; j < 8; ++j) inv[j] = 1.0f / lrun[j];
#pragma unroll
  for (int dt = 0; dt < 8; ++dt)
#pragma unroll
    for (int j = 0; j < 8; ++j) {
      int s = s0 + j + half * 8;
      attnb[((size_t)(b * 1024 + s)) * 4096 + h * 128 + dt * 16 + (lane & 15)] =
          (__bf16)(o[dt][j] * inv[j]);
    }
}

// ---------------- host launch ----------------
extern "C" void kernel_launch(void* const* d_in, const int* in_sizes, int n_in,
                              void* d_out, int out_size, void* d_ws, size_t ws_size,
                              hipStream_t stream) {
  (void)in_sizes; (void)n_in; (void)out_size; (void)ws_size;
  const float* hidden = (const float*)d_in[0];
  const float* w_qkv  = (const float*)d_in[1];
  const float* w_o    = (const float*)d_in[2];
  const float* cosb   = (const float*)d_in[3];
  const float* sinb   = (const float*)d_in[4];
  const float* kc     = (const float*)d_in[5];
  const float* vc     = (const float*)d_in[6];
  float* out = (float*)d_out;

  char* ws = (char*)d_ws;
  size_t off = 0;
  __bf16* hsb   = (__bf16*)(ws + off); off += (size_t)Mrows * H * 2;        // 16 MiB
  __bf16* wqkvT = (__bf16*)(ws + off); off += (size_t)NQKV * H * 2;         // 48 MiB
  __bf16* woT   = (__bf16*)(ws + off); off += (size_t)H * H * 2;            // 32 MiB
  __bf16* qkvt  = (__bf16*)(ws + off); off += (size_t)Mrows * NQKV * 2;     // 24 MiB
  __bf16* qb    = (__bf16*)(ws + off); off += (size_t)Bn * NQ * S * D * 2;  // 16 MiB
  __bf16* kall  = (__bf16*)(ws + off); off += (size_t)Bn * NKV * Tt * D * 2;//  8 MiB
  __bf16* vallT = (__bf16*)(ws + off); off += (size_t)Bn * NKV * Tt * D * 2;//  8 MiB
  __bf16* attnb = (__bf16*)(ws + off);                                      // 16 MiB

  k_cvt_bf16<<<(Mrows * H / 4 + 255) / 256, 256, 0, stream>>>(hidden, hsb, Mrows * H);
  k_transpose_cvt<<<dim3(NQKV / 32, H / 32), dim3(32, 8), 0, stream>>>(w_qkv, wqkvT, H, NQKV);
  k_transpose_cvt<<<dim3(H / 32, H / 32), dim3(32, 8), 0, stream>>>(w_o, woT, H, H);
  k_cvt_kv<<<(Bn * Pc * NKV * D) / 256, 256, 0, stream>>>(kc, vc, kall, vallT);
  k_gemm<1><<<dim3(NQKV / 128, Mrows / 128), 256, 0, stream>>>(hsb, wqkvT, qkvt,
                                                               Mrows, NQKV, H);
  k_rope_scatter<<<dim3(Mrows, 48), 128, 0, stream>>>(qkvt, cosb, sinb, qb, kall, vallT);
  k_attn<<<512, 256, 0, stream>>>(qb, kall, vallT, attnb);
  k_gemm<0><<<dim3(H / 128, Mrows / 128), 256, 0, stream>>>(attnb, woT, out,
                                                            Mrows, H, H);
}
